// Cell_1_0_35699768164365
// MI455X (gfx1250) — matrix operand order repaired and checked by reading, not compiled, not run on hardware
//
#include <hip/hip_runtime.h>

typedef __attribute__((ext_vector_type(16))) _Float16 v16h;
typedef __attribute__((ext_vector_type(8)))  float    v8f;

constexpr int BATCHN = 2048;
constexpr int CH     = 64;    // channels
constexpr int SAW    = 121;   // N*D
constexpr int SAP    = 128;   // padded SA
constexpr int HIDN   = 128;   // router hidden
constexpr int NPATH  = 4;

// ---------------- LDS (phase-aliased, exactly 160 KB) ----------------
struct __align__(64) Smem {
  union U0 { _Float16 Xl[CH*SAP]; float S1[CH*CH]; float bcs[CH]; } u0;     // 16 KB
  union U1 { _Float16 Xh[CH*SAP]; float S2[CH*CH]; } u1;                    // 16 KB
  union U2 {
    _Float16 W[SAP*SAP];                                                    // 32 KB
    struct { _Float16 P1[CH*CH]; _Float16 P2[CH*CH]; _Float16 Wc[CH*2*CH]; } p;
    struct { float pooled[2*CH]; float hidden[HIDN]; } r;
  } u2;
  union U3 {
    struct { _Float16 Q1[CH*SAP]; _Float16 Q2[CH*SAP]; } qq;                // 32 KB
    _Float16 EmbT[SAP*2*CH];
  } u3;
  union U4 { _Float16 K1[CH*SAP]; _Float16 G1[SAP*CH]; } u4;                // 16 KB
  union U5 { _Float16 K2[CH*SAP]; _Float16 G2[SAP*CH]; } u5;                // 16 KB
  _Float16 V1[CH*SAP];                                                      // 16 KB
  _Float16 V2[CH*SAP];                                                      // 16 KB
};

// A fragment: 16x32 f16 tile; lanes 0-15 rows M, kh=lane>>4 selects K sub-halves.
// Per-lane halves: K = kb + kh*16 + {0..15} (contiguous).
__device__ inline v16h load_frag_A(const _Float16* X, int stride, int mr, int kb, int lane) {
  int m  = lane & 15;
  int kh = lane >> 4;
  const _Float16* p = X + (mr + m) * stride + kb + kh * 16;
  union { v16h v; uint4 q[2]; } u;
  u.q[0] = *(const uint4*)p;
  u.q[1] = *(const uint4*)(p + 8);
  return u.v;
}

// B fragment: 32x16 f16 tile from n-major storage Bt[n][k] (stride in halves).
// Lane n = nb + (lane&15); halves j -> K = kb + (lane>>4)*16 + j (one 32B chunk).
__device__ inline v16h load_frag_B(const _Float16* Bt, int stride, int kb, int nb, int lane) {
  int n  = lane & 15;
  int kh = lane >> 4;
  const _Float16* p = Bt + (nb + n) * stride + kb + kh * 16;
  return *(const v16h*)p;
}

__device__ inline v8f wmma16(v16h a, v16h b, v8f c) {
  return __builtin_amdgcn_wmma_f32_16x16x32_f16(false, a, false, b, (short)0, c, false, false);
}

__global__ __launch_bounds__(256)
void fused_cross_sa_kernel(const float* __restrict__ lidar, const float* __restrict__ hsi,
                           const float* __restrict__ Wq, const float* __restrict__ bq,
                           const float* __restrict__ Wk, const float* __restrict__ bk,
                           const float* __restrict__ Wv, const float* __restrict__ bv,
                           const float* __restrict__ Wr1, const float* __restrict__ br1,
                           const float* __restrict__ Wr2, const float* __restrict__ br2,
                           const float* __restrict__ Wcv, const float* __restrict__ bc,
                           float* __restrict__ out_emb, float* __restrict__ out_pp) {
  __shared__ Smem sm;
  const int b    = blockIdx.x;
  const int tid  = threadIdx.x;
  const int lane = tid & 31;
  const int wave = tid >> 5;

  const float* lb = lidar + (size_t)b * CH * SAW;
  const float* hb = hsi   + (size_t)b * CH * SAW;

  // ---- Phase 0: stage inputs as f16 [CH][SAP], zero-padded ----
  for (int e = tid; e < CH * SAP; e += 256) {
    int r = e >> 7, s = e & 127;
    float lv = (s < SAW) ? lb[r * SAW + s] : 0.0f;
    float hv = (s < SAW) ? hb[r * SAW + s] : 0.0f;
    sm.u0.Xl[e] = (_Float16)lv;
    sm.u1.Xh[e] = (_Float16)hv;
  }
  __syncthreads();

  // ---- Router: pooled mean -> MLP -> softmax(4) (uses W-region union) ----
  if (tid < 2 * CH) {
    const float* src = (tid < CH) ? (lb + tid * SAW) : (hb + (tid - CH) * SAW);
    float s = 0.0f;
    for (int i = 0; i < SAW; ++i) s += src[i];
    sm.u2.r.pooled[tid] = s * (1.0f / (float)SAW);
  }
  __syncthreads();
  if (tid < HIDN) {
    float acc = br1[tid];
    for (int c = 0; c < 2 * CH; ++c) acc += sm.u2.r.pooled[c] * Wr1[c * HIDN + tid];
    sm.u2.r.hidden[tid] = acc > 0.0f ? acc : 0.0f;
  }
  __syncthreads();
  if (tid == 0) {
    float lg[NPATH];
    for (int p = 0; p < NPATH; ++p) {
      float acc = br2[p];
      for (int j = 0; j < HIDN; ++j) acc += sm.u2.r.hidden[j] * Wr2[j * NPATH + p];
      lg[p] = acc;
    }
    float mx = fmaxf(fmaxf(lg[0], lg[1]), fmaxf(lg[2], lg[3]));
    float sum = 0.0f;
    for (int p = 0; p < NPATH; ++p) { lg[p] = __expf(lg[p] - mx); sum += lg[p]; }
    float inv = 1.0f / sum;
    for (int p = 0; p < NPATH; ++p) out_pp[b * NPATH + p] = lg[p] * inv;
  }
  __syncthreads();

  // ---- Zero W region once (pad rows/cols stay zero for all 3 weights) ----
  for (int e = tid; e < SAP * SAP / 2; e += 256) ((unsigned int*)sm.u2.W)[e] = 0u;
  __syncthreads();

  // ---- Phase 1: q/k/v GEMMs, one weight at a time ----
  const float* Wg[3] = {Wq, Wk, Wv};
  const float* bg[3] = {bq, bk, bv};
  for (int wi = 0; wi < 3; ++wi) {
    const float* Wp = Wg[wi];
    for (int e = tid; e < SAW * SAW; e += 256) {       // W[s][t] row-major = n-major B
      int s = e / SAW, t2 = e - s * SAW;
      sm.u2.W[s * SAP + t2] = (_Float16)Wp[e];
    }
    __syncthreads();
    _Float16* d1 = (wi == 0) ? sm.u3.qq.Q1 : (wi == 1) ? sm.u4.K1 : sm.V1;
    _Float16* d2 = (wi == 0) ? sm.u3.qq.Q2 : (wi == 1) ? sm.u5.K2 : sm.V2;
    const _Float16* s1 = (wi == 2) ? sm.u1.Xh : sm.u0.Xl;  // v1 = h@Wv, v2 = l@Wv
    const _Float16* s2 = (wi == 2) ? sm.u0.Xl : sm.u1.Xh;
    const float* bias = bg[wi];
    for (int t = wave; t < 64; t += 8) {
      int inp = t >> 5, mi = (t >> 3) & 3, ni = t & 7;
      const _Float16* src = inp ? s2 : s1;
      _Float16* dst = inp ? d2 : d1;
      int mr = mi * 16, nb = ni * 16;
      v8f acc = {0.f,0.f,0.f,0.f,0.f,0.f,0.f,0.f};
      #pragma unroll
      for (int kk = 0; kk < 4; ++kk) {
        v16h a  = load_frag_A(src, SAP, mr, kk * 32, lane);
        v16h bf = load_frag_B(sm.u2.W, SAP, kk * 32, nb, lane);
        acc = wmma16(a, bf, acc);
      }
      int n = lane & 15, mh = lane >> 4;
      int col = nb + n;
      float bcol = (col < SAW) ? bias[col] : 0.0f;
      #pragma unroll
      for (int i = 0; i < 8; ++i) {
        int row = mr + mh * 8 + i;
        float v = (col < SAW) ? (acc[i] + bcol) : 0.0f;   // zero pad cols exactly
        dst[row * SAP + col] = (_Float16)v;
      }
    }
    __syncthreads();
  }

  // ---- Phase 2: scores S = Q @ K^T (K rows are n-major B) + Wc staging ----
  for (int t = wave; t < 32; t += 8) {
    int dir = t >> 4, mi = (t >> 2) & 3, ni = t & 3;
    const _Float16* Qd = dir ? sm.u3.qq.Q2 : sm.u3.qq.Q1;
    const _Float16* Kd = dir ? sm.u5.K2 : sm.u4.K1;
    float* Sd = dir ? sm.u1.S2 : sm.u0.S1;
    int mr = mi * 16, nb = ni * 16;
    v8f acc = {0.f,0.f,0.f,0.f,0.f,0.f,0.f,0.f};
    #pragma unroll
    for (int kk = 0; kk < 4; ++kk) {
      v16h a  = load_frag_A(Qd, SAP, mr, kk * 32, lane);
      v16h bf = load_frag_B(Kd, SAP, kk * 32, nb, lane);
      acc = wmma16(a, bf, acc);
    }
    int n = lane & 15, mh = lane >> 4;
    #pragma unroll
    for (int i = 0; i < 8; ++i) Sd[(mr + mh * 8 + i) * CH + nb + n] = acc[i];
  }
  for (int e = tid; e < CH * 2 * CH; e += 256) sm.u2.p.Wc[e] = (_Float16)Wcv[e];
  __syncthreads();

  // ---- Phase 3: softmax rows -> P (f16); gating G = (Q*V)^T (n-major) ----
  if (tid < 2 * CH) {
    int dir = tid >> 6, r = tid & 63;
    float* Sd = dir ? sm.u1.S2 : sm.u0.S1;
    _Float16* Pd = dir ? sm.u2.p.P2 : sm.u2.p.P1;
    float mx = -3.4e38f;
    for (int g = 0; g < CH; ++g) mx = fmaxf(mx, Sd[r * CH + g]);
    float sum = 0.0f;
    for (int g = 0; g < CH; ++g) sum += __expf(Sd[r * CH + g] - mx);
    float inv = 1.0f / sum;
    for (int g = 0; g < CH; ++g)
      Pd[r * CH + g] = (_Float16)(__expf(Sd[r * CH + g] - mx) * inv);
  }
  for (int e = tid; e < 2 * CH * SAP; e += 256) {      // G writes over dead K region
    int dir = e >> 13, idx = e & 8191;
    int r = idx >> 7, s = idx & 127;
    const _Float16* Qd = dir ? sm.u3.qq.Q2 : sm.u3.qq.Q1;
    const _Float16* Vd = dir ? sm.V2 : sm.V1;
    _Float16* Gd = dir ? sm.u5.G2 : sm.u4.G1;
    float gv = (float)Qd[idx] * (float)Vd[idx];
    Gd[s * CH + r] = (_Float16)gv;                     // transposed -> n-major B
  }
  __syncthreads();

  // ---- Phase 4: attn out = P @ G -> EmbT[s][c] (over dead Q region) ----
  if (tid < CH) sm.u0.bcs[tid] = bc[tid];              // S region dead now
  for (int t = wave; t < 64; t += 8) {
    int dir = t >> 5, mi = (t >> 3) & 3, ni = t & 7;
    const _Float16* Pd = dir ? sm.u2.p.P2 : sm.u2.p.P1;
    const _Float16* Gd = dir ? sm.u5.G2 : sm.u4.G1;
    int mr = mi * 16, nb = ni * 16;
    v8f acc = {0.f,0.f,0.f,0.f,0.f,0.f,0.f,0.f};
    #pragma unroll
    for (int kk = 0; kk < 2; ++kk) {
      v16h a  = load_frag_A(Pd, CH, mr, kk * 32, lane);
      v16h bf = load_frag_B(Gd, CH, kk * 32, nb, lane);
      acc = wmma16(a, bf, acc);
    }
    int n = lane & 15, mh = lane >> 4;
    int s = nb + n;
    int coff = dir ? 0 : CH;     // dir0 = h_emb -> channels 64..127; dir1 = l_emb -> 0..63
    #pragma unroll
    for (int i = 0; i < 8; ++i)
      sm.u3.EmbT[s * (2 * CH) + coff + mr + mh * 8 + i] = (_Float16)acc[i];
  }
  __syncthreads();

  // ---- Phase 5: 1x1 conv  emb = Wc @ cat + bc  (B = EmbT, already n-major) ----
  for (int t = wave; t < 32; t += 8) {
    int mi = t >> 3, ni = t & 7;
    int mr = mi * 16, nb = ni * 16;
    v8f acc = {0.f,0.f,0.f,0.f,0.f,0.f,0.f,0.f};
    #pragma unroll
    for (int kk = 0; kk < 4; ++kk) {
      v16h a  = load_frag_A(sm.u2.p.Wc, 2 * CH, mr, kk * 32, lane);
      v16h bf = load_frag_B(sm.u3.EmbT, 2 * CH, kk * 32, nb, lane);
      acc = wmma16(a, bf, acc);
    }
    int n = lane & 15, mh = lane >> 4;
    int col = nb + n;
    if (col < SAW) {
      #pragma unroll
      for (int i = 0; i < 8; ++i) {
        int row = mr + mh * 8 + i;
        out_emb[((size_t)b * CH + row) * SAW + col] = acc[i] + sm.u0.bcs[row];
      }
    }
  }
}

extern "C" void kernel_launch(void* const* d_in, const int* in_sizes, int n_in,
                              void* d_out, int out_size, void* d_ws, size_t ws_size,
                              hipStream_t stream) {
  (void)in_sizes; (void)n_in; (void)out_size; (void)d_ws; (void)ws_size;
  const float* lidar = (const float*)d_in[0];
  const float* hsi   = (const float*)d_in[1];
  const float* Wq    = (const float*)d_in[2];
  const float* bq    = (const float*)d_in[3];
  const float* Wk    = (const float*)d_in[4];
  const float* bk    = (const float*)d_in[5];
  const float* Wv    = (const float*)d_in[6];
  const float* bv    = (const float*)d_in[7];
  const float* Wr1   = (const float*)d_in[8];
  const float* br1   = (const float*)d_in[9];
  const float* Wr2   = (const float*)d_in[10];
  const float* br2   = (const float*)d_in[11];
  const float* Wcv   = (const float*)d_in[12];
  const float* bc    = (const float*)d_in[13];

  float* out_emb = (float*)d_out;
  float* out_pp  = out_emb + (size_t)BATCHN * CH * SAW;

  fused_cross_sa_kernel<<<BATCHN, 256, 0, stream>>>(
      lidar, hsi, Wq, bq, Wk, bk, Wv, bv, Wr1, br1, Wr2, br2, Wcv, bc,
      out_emb, out_pp);
}
